// MultiHeadAttention_50259707298028
// MI455X (gfx1250) — matrix operand order repaired and checked by reading, not compiled, not run on hardware
//
#include <hip/hip_runtime.h>

#define DM   2048
#define NH   16
#define DH   128
#define BATCH 2
#define SEQ  2048
#define MTOT (BATCH * SEQ)   // 4096
#define HE   (NH * DH)       // 2048

typedef __attribute__((ext_vector_type(16))) __bf16       v16bf;
typedef __attribute__((ext_vector_type(8)))  float        v8f;
typedef __attribute__((ext_vector_type(8)))  unsigned int v8u;

// ---------------- helpers ----------------

__device__ __forceinline__ unsigned short f2bf(float f) {
  unsigned int u = __builtin_bit_cast(unsigned int, f);
  u += 0x7FFFu + ((u >> 16) & 1u);              // round-to-nearest-even
  return (unsigned short)(u >> 16);
}

// A-fragment: 16(M) x 32(K) bf16, row-major source with row stride ld.
// ISA layout: lanes 0-15 / 16-31 both hold M = lane%16; VGPR v holds K pair
// base = (v>=4 ? 16 : 0) + 8*(lane/16) + 2*(v%4).
__device__ __forceinline__ v8u load_frag_a(const unsigned short* base, int ld) {
  const int lane = threadIdx.x & 31;
  const int m = lane & 15, g = lane >> 4;
  const unsigned short* r = base + (size_t)m * ld;
  v8u o;
#pragma unroll
  for (int v = 0; v < 8; ++v) {
    const int k = ((v & 4) << 2) + g * 8 + 2 * (v & 3);
    o[v] = *(const unsigned int*)(r + k);
  }
  return o;
}

// B-fragment: 32(K) x 16(N) bf16 loaded from B^T stored [N, K] row-major.
// ISA layout: lane holds N = lane%16; VGPR v holds K pair
// base = (v>=4 ? 16 : 0) + 8*(lane/16) + 2*(v%4).
__device__ __forceinline__ v8u load_frag_bt(const unsigned short* bt, int ld) {
  const int lane = threadIdx.x & 31;
  const int n = lane & 15, g = lane >> 4;
  const unsigned short* r = bt + (size_t)n * ld;
  v8u o;
#pragma unroll
  for (int v = 0; v < 8; ++v)
    o[v] = *(const unsigned int*)(r + ((v & 4) << 2) + g * 8 + 2 * (v & 3));
  return o;
}

__device__ __forceinline__ v8f wmma_bf16(v8u a, v8u b, v8f c) {
  return __builtin_amdgcn_wmma_f32_16x16x32_bf16(
      false, __builtin_bit_cast(v16bf, a),
      false, __builtin_bit_cast(v16bf, b),
      (short)0, c, false, false);
}

// Async copy 16B/lane from global memory into LDS (CDNA5 ASYNCcnt path).
__device__ __forceinline__ void async_ld_b128(unsigned int lds_off,
                                              const unsigned short* g) {
  asm volatile("global_load_async_to_lds_b128 %0, %1, off"
               :: "v"(lds_off), "v"(g) : "memory");
}

__device__ __forceinline__ unsigned int lds_off32(const void* p) {
  return (unsigned int)(size_t)p;
}

// ---------------- kernels ----------------

__global__ void mha_cast_bf16(const float* __restrict__ in,
                              unsigned short* __restrict__ out, int n) {
  int i = (blockIdx.x * blockDim.x + threadIdx.x) * 4;
  if (i < n) {  // all sizes are multiples of 4
    float4 f = *(const float4*)(in + i);
    out[i + 0] = f2bf(f.x);
    out[i + 1] = f2bf(f.y);
    out[i + 2] = f2bf(f.z);
    out[i + 3] = f2bf(f.w);
  }
}

// C[M=4096, N=2048] = A[M,K=2048] * Bt[N,K]^T, output scattered to
// [B,H,S,E] (transposeV=0) or [B,H,E,S] (transposeV=1), bf16, scaled.
// 128x128 block tile, double-buffered async-LDS staging of 128x32 subtiles.
__global__ __launch_bounds__(256)
void mha_gemm_proj(const unsigned short* __restrict__ A,
                   const unsigned short* __restrict__ Bt,
                   unsigned short* __restrict__ C,
                   float scale, int transposeV) {
  __shared__ unsigned short Atile[2][128 * 32];
  __shared__ unsigned short Btile[2][128 * 32];

  const int tid = threadIdx.x;
  const int wave = tid >> 5;
  const int m0b = blockIdx.x * 128;
  const int n0b = blockIdx.y * 128;
  const int mw = (wave & 3) * 32;
  const int nw = (wave >> 2) * 64;

  const int row0 = tid >> 2,          seg0 = (tid & 3) * 8;         // elems
  const int row1 = (tid + 256) >> 2,  seg1 = ((tid + 256) & 3) * 8;

  v8f acc[2][4];
#pragma unroll
  for (int i = 0; i < 2; ++i)
#pragma unroll
    for (int j = 0; j < 4; ++j)
#pragma unroll
      for (int r = 0; r < 8; ++r) acc[i][j][r] = 0.0f;

  // stage subtile k0 into buffer buf (4 async b128 per thread)
  auto stage = [&](int buf, int k0) {
    async_ld_b128(lds_off32(&Atile[buf][row0 * 32 + seg0]),
                  A + (size_t)(m0b + row0) * DM + k0 + seg0);
    async_ld_b128(lds_off32(&Atile[buf][row1 * 32 + seg1]),
                  A + (size_t)(m0b + row1) * DM + k0 + seg1);
    async_ld_b128(lds_off32(&Btile[buf][row0 * 32 + seg0]),
                  Bt + (size_t)(n0b + row0) * DM + k0 + seg0);
    async_ld_b128(lds_off32(&Btile[buf][row1 * 32 + seg1]),
                  Bt + (size_t)(n0b + row1) * DM + k0 + seg1);
  };

  stage(0, 0);
  const int TILES = DM / 32;
  for (int t = 0; t < TILES; ++t) {
    const int cur = t & 1;
    if (t + 1 < TILES) {
      stage(cur ^ 1, (t + 1) * 32);
      asm volatile("s_wait_asynccnt 0x4" ::: "memory");  // tile t arrived (ours)
    } else {
      asm volatile("s_wait_asynccnt 0x0" ::: "memory");
    }
    __syncthreads();  // tile t arrived (all waves)

    v8u a0 = load_frag_a(&Atile[cur][mw * 32], 32);
    v8u a1 = load_frag_a(&Atile[cur][(mw + 16) * 32], 32);
    v8u b[4];
#pragma unroll
    for (int j = 0; j < 4; ++j)
      b[j] = load_frag_bt(&Btile[cur][(nw + 16 * j) * 32], 32);
#pragma unroll
    for (int j = 0; j < 4; ++j) {
      acc[0][j] = wmma_bf16(a0, b[j], acc[0][j]);
      acc[1][j] = wmma_bf16(a1, b[j], acc[1][j]);
    }
    __syncthreads();  // done reading before buffer is overwritten
  }

  const int lane = threadIdx.x & 31;
  const int g = lane >> 4, cn = lane & 15;
#pragma unroll
  for (int mi = 0; mi < 2; ++mi)
#pragma unroll
    for (int nj = 0; nj < 4; ++nj)
#pragma unroll
      for (int r = 0; r < 8; ++r) {
        const int m = m0b + mw + 16 * mi + r + 8 * g;
        const int n = n0b + nw + 16 * nj + cn;
        const int b = m >> 11, s = m & (SEQ - 1);
        const int h = n >> 7,  e = n & (DH - 1);
        const unsigned short val = f2bf(acc[mi][nj][r] * scale);
        if (!transposeV)
          C[((size_t)(b * NH + h) * SEQ + s) * DH + e] = val;
        else
          C[((size_t)(b * NH + h) * DH + e) * SEQ + s] = val;
      }
}

// Flash attention: one wave owns 16 query rows x full head dim (E=128).
// Q pre-scaled by 1/sqrt(E). Streams KV in blocks of 32 with online softmax.
// No block barriers here (waves have different causal trip counts).
__global__ __launch_bounds__(128)
void mha_flash_attn(const unsigned short* __restrict__ Qb,  // [B,H,S,E]
                    const unsigned short* __restrict__ Kb,  // [B,H,S,E]
                    const unsigned short* __restrict__ Vt,  // [B,H,E,S]
                    unsigned short* __restrict__ Zb) {      // [B,H,S,E]
  __shared__ unsigned short pbuf[4 * 16 * 32];  // per-wave P-transpose slab

  const int wave = threadIdx.x >> 5;
  const int lane = threadIdx.x & 31;
  const int g = lane >> 4, hn = lane & 15;
  const int bh = blockIdx.y;
  const int qbase = (blockIdx.x * 4 + wave) * 16;

  const unsigned short* Qh = Qb + (size_t)bh * SEQ * DH;
  const unsigned short* Kh = Kb + (size_t)bh * SEQ * DH;
  const unsigned short* Vh = Vt + (size_t)bh * DH * SEQ;
  unsigned short* pw = pbuf + wave * 512;

  v8u qf[4];
#pragma unroll
  for (int c = 0; c < 4; ++c)
    qf[c] = load_frag_a(Qh + (size_t)qbase * DH + 32 * c, DH);

  v8f o[8];
  float mrow[8], lrow[8];
#pragma unroll
  for (int c = 0; c < 8; ++c)
#pragma unroll
    for (int r = 0; r < 8; ++r) o[c][r] = 0.0f;
#pragma unroll
  for (int r = 0; r < 8; ++r) { mrow[r] = -3.0e38f; lrow[r] = 0.0f; }

  for (int j = 0; j < qbase + 16; j += 32) {
    v8f s0, s1;
#pragma unroll
    for (int r = 0; r < 8; ++r) { s0[r] = 0.0f; s1[r] = 0.0f; }

#pragma unroll
    for (int c = 0; c < 4; ++c) {
      v8u b0 = load_frag_bt(Kh + (size_t)j * DH + 32 * c, DH);
      v8u b1 = load_frag_bt(Kh + (size_t)(j + 16) * DH + 32 * c, DH);
      s0 = wmma_bf16(qf[c], b0, s0);
      s1 = wmma_bf16(qf[c], b1, s1);
    }

    // causal mask + per-row block max (reduce across 16-lane half-groups)
    float bmax[8];
#pragma unroll
    for (int r = 0; r < 8; ++r) {
      const int qrow = qbase + r + 8 * g;
      if (j + hn > qrow)      s0[r] = -3.0e38f;
      if (j + 16 + hn > qrow) s1[r] = -3.0e38f;
      float v = fmaxf(s0[r], s1[r]);
      v = fmaxf(v, __shfl_xor(v, 1, 32));
      v = fmaxf(v, __shfl_xor(v, 2, 32));
      v = fmaxf(v, __shfl_xor(v, 4, 32));
      v = fmaxf(v, __shfl_xor(v, 8, 32));
      bmax[r] = v;
    }

    // online softmax: rescale, exponentiate, accumulate row sums, stash P
#pragma unroll
    for (int r = 0; r < 8; ++r) {
      const float mnew  = fmaxf(mrow[r], bmax[r]);
      const float alpha = __expf(mrow[r] - mnew);
      const float p0 = __expf(s0[r] - mnew);
      const float p1 = __expf(s1[r] - mnew);
      float sum = p0 + p1;
      sum += __shfl_xor(sum, 1, 32);
      sum += __shfl_xor(sum, 2, 32);
      sum += __shfl_xor(sum, 4, 32);
      sum += __shfl_xor(sum, 8, 32);
      lrow[r] = lrow[r] * alpha + sum;
      mrow[r] = mnew;
#pragma unroll
      for (int c2 = 0; c2 < 8; ++c2) o[c2][r] *= alpha;
      const int row = r + 8 * g;
      pw[row * 32 + hn]      = f2bf(p0);
      pw[row * 32 + 16 + hn] = f2bf(p1);
    }
    __asm__ volatile("s_wait_dscnt 0" ::: "memory");

    // reload P (16x32) as an A-fragment
    v8u pa;
#pragma unroll
    for (int v = 0; v < 8; ++v) {
      const int k = ((v & 4) << 2) + g * 8 + 2 * (v & 3);
      pa[v] = *(const unsigned int*)(pw + hn * 32 + k);
    }

    // O += P * V   (V^T stored [E,S] -> contiguous B-fragment loads)
#pragma unroll
    for (int c2 = 0; c2 < 8; ++c2) {
      v8u bv = load_frag_bt(Vh + (size_t)(16 * c2) * SEQ + j, SEQ);
      o[c2] = wmma_bf16(pa, bv, o[c2]);
    }
  }

  // normalize and write Z (bf16)
#pragma unroll
  for (int c2 = 0; c2 < 8; ++c2)
#pragma unroll
    for (int r = 0; r < 8; ++r) {
      const int s = qbase + r + 8 * g;
      const int e = 16 * c2 + hn;
      Zb[((size_t)bh * SEQ + s) * DH + e] = f2bf(o[c2][r] / lrow[r]);
    }
}

// out[M=4096, N=2048] = Z[B,H,S,E] (viewed [M, K=H*E]) * Wo[h,d,e]^T, fp32 out.
// Same async-LDS double-buffered staging as the projection GEMM.
__global__ __launch_bounds__(256)
void mha_gemm_out(const unsigned short* __restrict__ Zb,
                  const unsigned short* __restrict__ Wo,   // [H, DM, E] bf16
                  float* __restrict__ out) {
  __shared__ unsigned short Atile[2][128 * 32];
  __shared__ unsigned short Btile[2][128 * 32];

  const int tid = threadIdx.x;
  const int wave = tid >> 5;
  const int m0b = blockIdx.x * 128;
  const int n0b = blockIdx.y * 128;
  const int mw = (wave & 3) * 32;
  const int nw = (wave >> 2) * 64;

  const int row0 = tid >> 2,          seg0 = (tid & 3) * 8;
  const int row1 = (tid + 256) >> 2,  seg1 = ((tid + 256) & 3) * 8;

  v8f acc[2][4];
#pragma unroll
  for (int i = 0; i < 2; ++i)
#pragma unroll
    for (int j = 0; j < 4; ++j)
#pragma unroll
      for (int r = 0; r < 8; ++r) acc[i][j][r] = 0.0f;

  auto stage = [&](int buf, int k0) {
    const int h = k0 >> 7, eb = k0 & (DH - 1);
    {
      const int m = m0b + row0, b = m >> 11, s = m & (SEQ - 1);
      async_ld_b128(lds_off32(&Atile[buf][row0 * 32 + seg0]),
                    Zb + ((size_t)(b * NH + h) * SEQ + s) * DH + eb + seg0);
    }
    {
      const int m = m0b + row1, b = m >> 11, s = m & (SEQ - 1);
      async_ld_b128(lds_off32(&Atile[buf][row1 * 32 + seg1]),
                    Zb + ((size_t)(b * NH + h) * SEQ + s) * DH + eb + seg1);
    }
    async_ld_b128(lds_off32(&Btile[buf][row0 * 32 + seg0]),
                  Wo + ((size_t)h * DM + n0b + row0) * DH + eb + seg0);
    async_ld_b128(lds_off32(&Btile[buf][row1 * 32 + seg1]),
                  Wo + ((size_t)h * DM + n0b + row1) * DH + eb + seg1);
  };

  stage(0, 0);
  const int TILES = HE / 32;
  for (int t = 0; t < TILES; ++t) {
    const int cur = t & 1;
    if (t + 1 < TILES) {
      stage(cur ^ 1, (t + 1) * 32);
      asm volatile("s_wait_asynccnt 0x4" ::: "memory");
    } else {
      asm volatile("s_wait_asynccnt 0x0" ::: "memory");
    }
    __syncthreads();

    v8u a0 = load_frag_a(&Atile[cur][mw * 32], 32);
    v8u a1 = load_frag_a(&Atile[cur][(mw + 16) * 32], 32);
    v8u b[4];
#pragma unroll
    for (int j = 0; j < 4; ++j)
      b[j] = load_frag_bt(&Btile[cur][(nw + 16 * j) * 32], 32);
#pragma unroll
    for (int j = 0; j < 4; ++j) {
      acc[0][j] = wmma_bf16(a0, b[j], acc[0][j]);
      acc[1][j] = wmma_bf16(a1, b[j], acc[1][j]);
    }
    __syncthreads();
  }

  const int lane = threadIdx.x & 31;
  const int g = lane >> 4, cn = lane & 15;
#pragma unroll
  for (int mi = 0; mi < 2; ++mi)
#pragma unroll
    for (int nj = 0; nj < 4; ++nj)
#pragma unroll
      for (int r = 0; r < 8; ++r) {
        const int m = m0b + mw + 16 * mi + r + 8 * g;
        const int n = n0b + nw + 16 * nj + cn;
        out[(size_t)m * DM + n] = acc[mi][nj][r];
      }
}

// ---------------- launcher ----------------

extern "C" void kernel_launch(void* const* d_in, const int* in_sizes, int n_in,
                              void* d_out, int out_size, void* d_ws, size_t ws_size,
                              hipStream_t stream) {
  (void)in_sizes; (void)n_in; (void)out_size; (void)ws_size;
  const float* x  = (const float*)d_in[0];
  const float* wq = (const float*)d_in[1];
  const float* wk = (const float*)d_in[2];
  const float* wv = (const float*)d_in[3];
  const float* wo = (const float*)d_in[4];
  float* out = (float*)d_out;

  unsigned short* ws = (unsigned short*)d_ws;
  size_t off = 0;
  unsigned short* xb  = ws + off; off += (size_t)MTOT * DM;
  unsigned short* wqb = ws + off; off += (size_t)HE * DM;
  unsigned short* wkb = ws + off; off += (size_t)HE * DM;
  unsigned short* wvb = ws + off; off += (size_t)HE * DM;
  unsigned short* wob = ws + off; off += (size_t)NH * DM * DH;
  unsigned short* Qb  = ws + off; off += (size_t)BATCH * NH * SEQ * DH;
  unsigned short* Kb  = ws + off; off += (size_t)BATCH * NH * SEQ * DH;
  unsigned short* Vt  = ws + off; off += (size_t)BATCH * NH * SEQ * DH;
  unsigned short* Zb  = ws + off; off += (size_t)BATCH * NH * SEQ * DH;

  {
    int n;
    n = MTOT * DM;     mha_cast_bf16<<<dim3((n / 4 + 255) / 256), 256, 0, stream>>>(x,  xb,  n);
    n = HE * DM;       mha_cast_bf16<<<dim3((n / 4 + 255) / 256), 256, 0, stream>>>(wq, wqb, n);
    n = HE * DM;       mha_cast_bf16<<<dim3((n / 4 + 255) / 256), 256, 0, stream>>>(wk, wkb, n);
    n = HE * DM;       mha_cast_bf16<<<dim3((n / 4 + 255) / 256), 256, 0, stream>>>(wv, wvb, n);
    n = NH * DM * DH;  mha_cast_bf16<<<dim3((n / 4 + 255) / 256), 256, 0, stream>>>(wo, wob, n);
  }

  const dim3 gp(MTOT / 128, HE / 128);
  const float qscale = 0.08838834764831845f;  // 1/sqrt(128)
  mha_gemm_proj<<<gp, 256, 0, stream>>>(xb, wqb, Qb, qscale, 0);
  mha_gemm_proj<<<gp, 256, 0, stream>>>(xb, wkb, Kb, 1.0f,   0);
  mha_gemm_proj<<<gp, 256, 0, stream>>>(xb, wvb, Vt, 1.0f,   1);

  mha_flash_attn<<<dim3(SEQ / 64, BATCH * NH), 128, 0, stream>>>(Qb, Kb, Vt, Zb);

  mha_gemm_out<<<dim3(MTOT / 128, DM / 128), 256, 0, stream>>>(Zb, wob, out);
}
